// TFTDeepSeekMoE_73735998537983
// MI455X (gfx1250) — matrix operand order repaired and checked by reading, not compiled, not run on hardware
//
#include <hip/hip_runtime.h>
#include <math.h>

// ---------------- problem constants (from reference setup_inputs) -----------
#define NTOK   1024          // B*S tokens
#define DMODEL 2048
#define NEXP   16
#define IDIM   1408
#define ISHD   2816
#define TOPK   6

// ---------------- tiling ----------------------------------------------------
#define BM    64
#define BN    64
#define BKO   64             // LDS K-depth per load phase (2 WMMA K-steps)
#define LSTR  72             // ushorts per LDS row (64 data + 8 pad, 16B-aligned)

typedef __attribute__((ext_vector_type(16))) __bf16 v16bf;
typedef __attribute__((ext_vector_type(2)))  __bf16 v2bf;
typedef __attribute__((ext_vector_type(8)))  float  v8f;

union Frag { uint4 q[2]; v16bf v; };

static __device__ __forceinline__ unsigned short bf16rne(float f) {
  unsigned u = __float_as_uint(f);
  u += 0x7FFFu + ((u >> 16) & 1u);           // round-to-nearest-even
  return (unsigned short)(u >> 16);
}
static __device__ __forceinline__ unsigned pack2(float a, float b) {
#if __has_builtin(__builtin_amdgcn_cvt_pk_bf16_f32)
  union { v2bf v; unsigned u; } cv;
  cv.v = __builtin_amdgcn_cvt_pk_bf16_f32(a, b);
  return cv.u;
#else
  return (unsigned)bf16rne(a) | ((unsigned)bf16rne(b) << 16);
#endif
}

// LDS byte offset of a generic pointer into __shared__ (ISA: flat LDS aperture
// keeps the LDS offset in addr[31:0]).
static __device__ __forceinline__ unsigned lds_off(const void* p) {
  return (unsigned)(uintptr_t)p;
}
// CDNA5 async copy: 16B global -> LDS per lane, tracked by ASYNCcnt.
static __device__ __forceinline__ void async_b128(unsigned ldsoff, const void* gptr) {
  asm volatile("global_load_async_to_lds_b128 %0, %1, off"
               :: "v"(ldsoff), "v"(gptr) : "memory");
}
static __device__ __forceinline__ void wait_async0() {
  asm volatile("s_wait_asynccnt 0x0" ::: "memory");
}

// ---------------- kernel 1: out = xe, xbf = bf16(x), zero counts ------------
__global__ __launch_bounds__(256) void init_kernel(
    const float* __restrict__ x, const float* __restrict__ xe,
    float* __restrict__ out, unsigned short* __restrict__ xbf,
    int* __restrict__ counts, int* __restrict__ offsets) {
  size_t i = ((size_t)blockIdx.x * 256 + threadIdx.x) * 4;
  if (i < (size_t)NTOK * DMODEL) {
    float4 v  = *(const float4*)(x + i);
    float4 e4 = *(const float4*)(xe + i);
    *(float4*)(out + i) = e4;
    uint2 p; p.x = pack2(v.x, v.y); p.y = pack2(v.z, v.w);
    *(uint2*)(xbf + i) = p;
  }
  if (blockIdx.x == 0 && threadIdx.x < NEXP + 1) {
    if (threadIdx.x < NEXP) counts[threadIdx.x] = 0;
    offsets[threadIdx.x] = 0;
  }
}

// ---------------- kernel 2: gate = softmax + top-6 + token lists ------------
__global__ __launch_bounds__(256) void gate_kernel(
    const float* __restrict__ x, const float* __restrict__ gw,
    float* __restrict__ cw, int* __restrict__ counts, int* __restrict__ tokens) {
  const int n = blockIdx.x;
  const int t = threadIdx.x;
  __shared__ float part[256];
  __shared__ float sc[NEXP];
  const int e = t >> 4, j = t & 15;          // 16 lanes per expert
  const float* xr = x  + (size_t)n * DMODEL;
  const float* wr = gw + (size_t)e * DMODEL;
  float acc = 0.f;
  for (int d = j; d < DMODEL; d += 16) acc += xr[d] * wr[d];
  part[t] = acc;
  __syncthreads();
  if (t < NEXP) {
    float s = 0.f;
    for (int q = 0; q < 16; ++q) s += part[t * 16 + q];
    sc[t] = s;
  }
  __syncthreads();
  if (t == 0) {
    float mx = sc[0];
    for (int i = 1; i < NEXP; ++i) mx = fmaxf(mx, sc[i]);
    float p[NEXP], sum = 0.f;
    for (int i = 0; i < NEXP; ++i) { p[i] = __expf(sc[i] - mx); sum += p[i]; }
    float inv = 1.0f / sum;
    float sel[NEXP], tmp[NEXP];
    for (int i = 0; i < NEXP; ++i) { p[i] *= inv; tmp[i] = p[i]; sel[i] = 0.f; }
    for (int k = 0; k < TOPK; ++k) {
      int am = 0; float bv = tmp[0];
      for (int i = 1; i < NEXP; ++i) if (tmp[i] > bv) { bv = tmp[i]; am = i; }
      sel[am] = p[am];                       // ROUTE_SCALE == 1.0
      tmp[am] = -1.0f;
      int slot = atomicAdd(&counts[am], 1);
      tokens[(size_t)am * NTOK + slot] = n;
    }
    for (int i = 0; i < NEXP; ++i) cw[(size_t)n * NEXP + i] = sel[i];
  }
}

// ---------------- kernel 3: 64-padded exclusive scan of counts --------------
__global__ void scan_kernel(const int* __restrict__ counts, int* __restrict__ offsets) {
  if (blockIdx.x == 0 && threadIdx.x == 0) {
    int off = 0;
    for (int e = 0; e < NEXP; ++e) { offsets[e] = off; off += (counts[e] + 63) & ~63; }
    offsets[NEXP] = off;
  }
}

// ---------------- kernel 4: fused up-proj NT-GEMM + SwiGLU ------------------
// G[row, i] = bf16( silu(X @ W1^T) * (X @ W3^T) ), X rows gathered per expert.
__global__ __launch_bounds__(128) void up_kernel(
    const unsigned short* __restrict__ Xbf,            // [NTOK, DMODEL] bf16
    const float* __restrict__ W1, const float* __restrict__ W3,
    size_t wstride,                                    // per-expert elements
    const int* __restrict__ tokens, const int* __restrict__ counts,
    const int* __restrict__ offsets,
    unsigned short* __restrict__ Gout, int Idim) {
  const int e   = blockIdx.z;
  const int cnt = counts ? counts[e] : NTOK;
  const int m0  = blockIdx.y * BM;
  if (m0 >= cnt) return;
  const int n0  = blockIdx.x * BN;
  const float* w1 = W1 + (size_t)e * wstride;
  const float* w3 = W3 + (size_t)e * wstride;
  const int rowbase = offsets ? offsets[e] : 0;
  const int* tl = tokens ? tokens + (size_t)e * NTOK : nullptr;

  __shared__ unsigned short lA [BM * LSTR];
  __shared__ unsigned short lB1[BN * LSTR];
  __shared__ unsigned short lB3[BN * LSTR];

  const int t = threadIdx.x;
  // A-tile: 64 rows x 64 K (bf16) = 512 x 16B chunks / 128 thr = 4 each,
  // copied global->LDS with async-DMA (no VGPR round trip).
  const unsigned short* asrc[4]; unsigned aoff[4];
  for (int q = 0; q < 4; ++q) {
    int c = t + q * 128;
    int row = c >> 3, part = c & 7;
    int s  = m0 + row;
    int sv = s < cnt ? s : cnt - 1;                    // clamp padded rows
    int tok = tl ? tl[sv] : sv;
    asrc[q] = Xbf + (size_t)tok * DMODEL + part * 8;
    aoff[q] = lds_off(&lA[row * LSTR + part * 8]);
  }
  const int lane = t & 31, wave = t >> 5;
  const int wm = wave >> 1, wn = wave & 1;             // 2x2 waves -> 32x32 each
  const int hi = lane >> 4, l16 = lane & 15;

  v8f zero = {0.f,0.f,0.f,0.f,0.f,0.f,0.f,0.f};
  v8f ch[2][2], cg[2][2];
  for (int i = 0; i < 2; ++i)
    for (int j = 0; j < 2; ++j) { ch[i][j] = zero; cg[i][j] = zero; }

  for (int k0 = 0; k0 < DMODEL; k0 += BKO) {
    for (int q = 0; q < 4; ++q) async_b128(aoff[q], asrc[q] + k0);
    // W1/W3 tiles: 64 rows x 64 K f32 = 1024 float4 chunks / 128 thr = 8 each.
    // Two batches of 4 with distinct temps so global loads clause together.
    for (int h = 0; h < 2; ++h) {
      float4 ta[4], tb[4]; int lrow[4], lp[4];
      for (int q = 0; q < 4; ++q) {
        int c = t + (h * 4 + q) * 128;
        int row = c >> 4, p = c & 15;
        lrow[q] = row; lp[q] = p;
        size_t gi = (size_t)(n0 + row) * DMODEL + k0 + p * 4;
        ta[q] = *(const float4*)(w1 + gi);
        tb[q] = *(const float4*)(w3 + gi);
        if (k0 + BKO < DMODEL) {
          __builtin_prefetch(w1 + gi + BKO, 0, 1);
          __builtin_prefetch(w3 + gi + BKO, 0, 1);
        }
      }
      for (int q = 0; q < 4; ++q) {
        uint2 pa; pa.x = pack2(ta[q].x, ta[q].y); pa.y = pack2(ta[q].z, ta[q].w);
        uint2 pb; pb.x = pack2(tb[q].x, tb[q].y); pb.y = pack2(tb[q].z, tb[q].w);
        *(uint2*)&lB1[lrow[q] * LSTR + lp[q] * 4] = pa;
        *(uint2*)&lB3[lrow[q] * LSTR + lp[q] * 4] = pb;
      }
    }
    wait_async0();
    __syncthreads();
    for (int kk = 0; kk < BKO; kk += 32) {
      Frag a[2], b1[2], b3[2];
      for (int i = 0; i < 2; ++i) {                    // A: ISA 16-bit 16x32 layout
        const unsigned short* pA = &lA[(wm * 32 + i * 16 + l16) * LSTR + kk];
        a[i].q[0] = *(const uint4*)(pA + hi * 8);      // K = 8*hi .. +7
        a[i].q[1] = *(const uint4*)(pA + 16 + hi * 8); // K = 16+8*hi .. +7
      }
      for (int j = 0; j < 2; ++j) {                    // B: ISA 16-bit layout (same K order as A)
        const unsigned short* p1 = &lB1[(wn * 32 + j * 16 + l16) * LSTR + kk];
        b1[j].q[0] = *(const uint4*)(p1 + hi * 8);
        b1[j].q[1] = *(const uint4*)(p1 + 16 + hi * 8);
        const unsigned short* p3 = &lB3[(wn * 32 + j * 16 + l16) * LSTR + kk];
        b3[j].q[0] = *(const uint4*)(p3 + hi * 8);
        b3[j].q[1] = *(const uint4*)(p3 + 16 + hi * 8);
      }
      for (int i = 0; i < 2; ++i)
        for (int j = 0; j < 2; ++j) {
          ch[i][j] = __builtin_amdgcn_wmma_f32_16x16x32_bf16(
              false, a[i].v, false, b1[j].v, (short)0, ch[i][j], false, false);
          cg[i][j] = __builtin_amdgcn_wmma_f32_16x16x32_bf16(
              false, a[i].v, false, b3[j].v, (short)0, cg[i][j], false, false);
        }
    }
    __syncthreads();
  }
  // epilogue: C layout M = r + 8*hi, N = l16
  for (int i = 0; i < 2; ++i)
    for (int j = 0; j < 2; ++j)
      for (int r = 0; r < 8; ++r) {
        int s   = m0 + wm * 32 + i * 16 + r + hi * 8;
        int col = n0 + wn * 32 + j * 16 + l16;
        float h = ch[i][j][r];
        float g = cg[i][j][r];
        float v = (h / (1.0f + __expf(-h))) * g;       // silu(h) * g
        Gout[(size_t)(rowbase + s) * Idim + col] = bf16rne(v);
      }
}

// ---------------- kernel 5: down-proj NT-GEMM + scaled accumulate -----------
// out[token, d] += scale * (G @ W2^T). One launch per expert => no atomics.
__global__ __launch_bounds__(128) void down_kernel(
    const unsigned short* __restrict__ G,              // [rows, Kd] bf16
    const float* __restrict__ W2, int e, int Kd,       // W2 + e*DMODEL*Kd: [DMODEL,Kd]
    const int* __restrict__ tokens, const int* __restrict__ counts,
    const int* __restrict__ offsets, const float* __restrict__ cw,
    float* __restrict__ out) {
  const int cnt = counts ? counts[e] : NTOK;
  const int m0  = blockIdx.y * BM;
  if (m0 >= cnt) return;
  const int n0  = blockIdx.x * BN;                     // output model-dim tile
  const float* w2 = W2 + (size_t)e * DMODEL * (size_t)Kd;
  const int rowbase = offsets ? offsets[e] : 0;
  const int* tl = tokens ? tokens + (size_t)e * NTOK : nullptr;

  __shared__ unsigned short lA[BM * LSTR];
  __shared__ unsigned short lB[BN * LSTR];

  const int t = threadIdx.x;
  const unsigned short* asrc[4]; unsigned aoff[4];
  for (int q = 0; q < 4; ++q) {
    int c = t + q * 128;
    int row = c >> 3, part = c & 7;
    asrc[q] = G + (size_t)(rowbase + m0 + row) * Kd + part * 8;  // padded region valid
    aoff[q] = lds_off(&lA[row * LSTR + part * 8]);
  }
  const int lane = t & 31, wave = t >> 5;
  const int wm = wave >> 1, wn = wave & 1;
  const int hi = lane >> 4, l16 = lane & 15;

  v8f zero = {0.f,0.f,0.f,0.f,0.f,0.f,0.f,0.f};
  v8f cacc[2][2];
  for (int i = 0; i < 2; ++i) for (int j = 0; j < 2; ++j) cacc[i][j] = zero;

  for (int k0 = 0; k0 < Kd; k0 += BKO) {
    for (int q = 0; q < 4; ++q) async_b128(aoff[q], asrc[q] + k0);
    for (int h = 0; h < 2; ++h) {
      float4 ta[4]; int lrow[4], lp[4];
      for (int q = 0; q < 4; ++q) {
        int c = t + (h * 4 + q) * 128;
        int row = c >> 4, p = c & 15;
        lrow[q] = row; lp[q] = p;
        size_t gi = (size_t)(n0 + row) * Kd + k0 + p * 4;
        ta[q] = *(const float4*)(w2 + gi);
        if (k0 + BKO < Kd) __builtin_prefetch(w2 + gi + BKO, 0, 1);
      }
      for (int q = 0; q < 4; ++q) {
        uint2 pa; pa.x = pack2(ta[q].x, ta[q].y); pa.y = pack2(ta[q].z, ta[q].w);
        *(uint2*)&lB[lrow[q] * LSTR + lp[q] * 4] = pa;
      }
    }
    wait_async0();
    __syncthreads();
    for (int kk = 0; kk < BKO; kk += 32) {
      Frag a[2], b[2];
      for (int i = 0; i < 2; ++i) {
        const unsigned short* pA = &lA[(wm * 32 + i * 16 + l16) * LSTR + kk];
        a[i].q[0] = *(const uint4*)(pA + hi * 8);
        a[i].q[1] = *(const uint4*)(pA + 16 + hi * 8);
      }
      for (int j = 0; j < 2; ++j) {
        const unsigned short* pB = &lB[(wn * 32 + j * 16 + l16) * LSTR + kk];
        b[j].q[0] = *(const uint4*)(pB + hi * 8);
        b[j].q[1] = *(const uint4*)(pB + 16 + hi * 8);
      }
      for (int i = 0; i < 2; ++i)
        for (int j = 0; j < 2; ++j)
          cacc[i][j] = __builtin_amdgcn_wmma_f32_16x16x32_bf16(
              false, a[i].v, false, b[j].v, (short)0, cacc[i][j], false, false);
    }
    __syncthreads();
  }
  for (int i = 0; i < 2; ++i)
    for (int j = 0; j < 2; ++j)
      for (int r = 0; r < 8; ++r) {
        int s = m0 + wm * 32 + i * 16 + r + hi * 8;
        if (s < cnt) {
          int tok = tl ? tl[s] : s;
          float scale = cw ? cw[(size_t)tok * NEXP + e] : 1.0f;
          int col = n0 + wn * 32 + j * 16 + l16;
          size_t oi = (size_t)tok * DMODEL + col;
          out[oi] += scale * cacc[i][j][r];            // race-free: serialized launches
        }
      }
}

// ---------------- launch -----------------------------------------------------
extern "C" void kernel_launch(void* const* d_in, const int* in_sizes, int n_in,
                              void* d_out, int out_size, void* d_ws, size_t ws_size,
                              hipStream_t stream) {
  (void)in_sizes; (void)n_in; (void)out_size; (void)ws_size;
  const float* x   = (const float*)d_in[0];
  const float* xe  = (const float*)d_in[1];
  const float* gw  = (const float*)d_in[2];
  const float* w1  = (const float*)d_in[3];
  const float* w2  = (const float*)d_in[4];
  const float* w3  = (const float*)d_in[5];
  const float* sw1 = (const float*)d_in[6];
  const float* sw2 = (const float*)d_in[7];
  const float* sw3 = (const float*)d_in[8];
  float* out = (float*)d_out;
  char*  ws  = (char*)d_ws;

  // workspace layout (all 256B aligned); total ~33.2 MB
  float*          cw      = (float*)(ws + 0);                 // [NTOK, NEXP]
  int*            counts  = (int*)  (ws + 65536);             // [NEXP]
  int*            offsets = (int*)  (ws + 65792);             // [NEXP+1]
  int*            tokens  = (int*)  (ws + 66048);             // [NEXP, NTOK]
  unsigned short* xbf     = (unsigned short*)(ws + 131584);   // [NTOK, DMODEL] bf16
  unsigned short* shb     = (unsigned short*)(ws + 4325888);  // [NTOK, ISHD]  bf16
  unsigned short* gbuf    = (unsigned short*)(ws + 10093056); // [<=8192, IDIM] bf16

  init_kernel<<<(NTOK * DMODEL / 4 + 255) / 256, 256, 0, stream>>>(
      x, xe, out, xbf, counts, offsets);
  gate_kernel<<<NTOK, 256, 0, stream>>>(x, gw, cw, counts, tokens);
  scan_kernel<<<1, 32, 0, stream>>>(counts, offsets);

  // shared experts: SHbf = silu(x@sw1^T) * (x@sw3^T)
  up_kernel<<<dim3(ISHD / BN, NTOK / BM, 1), 128, 0, stream>>>(
      xbf, sw1, sw3, (size_t)0, nullptr, nullptr, nullptr, shb, ISHD);
  // routed experts (gathered tokens, 2.7x less work than dense reference)
  up_kernel<<<dim3(IDIM / BN, NTOK / BM, NEXP), 128, 0, stream>>>(
      xbf, w1, w3, (size_t)IDIM * DMODEL, tokens, counts, offsets, gbuf, IDIM);

  // shared down-proj: out += SH @ sw2^T
  down_kernel<<<dim3(DMODEL / BN, NTOK / BM), 128, 0, stream>>>(
      shb, sw2, 0, ISHD, nullptr, nullptr, nullptr, nullptr, out);
  // routed down-proj: out += cw[tok,e] * (G_e @ w2_e^T), one launch per expert
  for (int e = 0; e < NEXP; ++e)
    down_kernel<<<dim3(DMODEL / BN, NTOK / BM), 128, 0, stream>>>(
        gbuf, w2, e, IDIM, tokens, counts, offsets, cw, out);
}
